// Job_Actor_61014305407240
// MI455X (gfx1250) — matrix operand order repaired and checked by reading, not compiled, not run on hardware
//
#include <hip/hip_runtime.h>
#include <hip/hip_bf16.h>
#include <math.h>

typedef _Float16 v16h __attribute__((ext_vector_type(16)));
typedef _Float16 v8h  __attribute__((ext_vector_type(8)));
typedef _Float16 v4h  __attribute__((ext_vector_type(4)));
typedef float    v8f  __attribute__((ext_vector_type(8)));

#define Bb   64
#define NN   600
#define NJ   30
#define NM   20
#define HH   128
#define RROWS (Bb * NN)          // 38400 total nodes
#define AROWS (Bb * NJ)          // 1920 actor rows
#define EPS  1e-5f

// ---------------------------------------------------------------------------
// Layer 1 fused: pooled = adj @ x (K=2), z = pooled @ W1_0  (bias absorbed by BN)
// ---------------------------------------------------------------------------
__global__ void k_layer1(const float* __restrict__ adj,   // [B,N,N]
                         const float* __restrict__ x,     // [B,N,2]
                         const float* __restrict__ W1,    // [2,H]
                         float* __restrict__ z)           // [B*N,H]
{
    __shared__ float r0[128], r1[128];
    const int node = blockIdx.x;
    const int b = node / NN, n = node % NN;
    const int t = threadIdx.x;
    const float* arow = adj + ((size_t)b * NN + n) * NN;
    const float* xb   = x + (size_t)b * NN * 2;
    float s0 = 0.f, s1 = 0.f;
    for (int m = t; m < NN; m += 128) {
        float a = arow[m];
        s0 += a * xb[m * 2 + 0];
        s1 += a * xb[m * 2 + 1];
    }
    r0[t] = s0; r1[t] = s1;
    __syncthreads();
    for (int s = 64; s > 0; s >>= 1) {
        if (t < s) { r0[t] += r0[t + s]; r1[t] += r1[t + s]; }
        __syncthreads();
    }
    float p0 = r0[0], p1 = r1[0];
    z[(size_t)node * HH + t] = p0 * W1[t] + p1 * W1[HH + t];
}

// ---------------------------------------------------------------------------
// BN stats helpers
// ---------------------------------------------------------------------------
__global__ void k_zero_stats(float* __restrict__ stats) {
    if (threadIdx.x < 256) stats[threadIdx.x] = 0.f;
}

__global__ void k_stats(const float* __restrict__ z, float* __restrict__ stats) {
    const int c = threadIdx.x & 127;
    const int rh = threadIdx.x >> 7;
    const int rstart = blockIdx.x * 2 + rh;
    const int rstep  = gridDim.x * 2;
    float s = 0.f, s2 = 0.f;
    for (int r = rstart; r < RROWS; r += rstep) {
        float v = z[(size_t)r * HH + c];
        s += v; s2 += v * v;
    }
    atomicAdd(&stats[c], s);
    atomicAdd(&stats[128 + c], s2);
}

__global__ void k_norm_relu(float* __restrict__ z, const float* __restrict__ stats) {
    const float inv_cnt = 1.0f / (float)RROWS;
    const size_t total = (size_t)RROWS * HH;
    for (size_t i = (size_t)blockIdx.x * blockDim.x + threadIdx.x; i < total;
         i += (size_t)gridDim.x * blockDim.x) {
        int c = (int)(i & (HH - 1));
        float m = stats[c] * inv_cnt;
        float v = stats[128 + c] * inv_cnt - m * m;
        float h = (z[i] - m) * rsqrtf(v + EPS);
        z[i] = h > 0.f ? h : 0.f;
    }
}

// ---------------------------------------------------------------------------
// WMMA GEMM: C[M,N] = act(A[M,K] * Bw[K,N] + bias), f32 in/out, f16 compute.
// 128 threads = 4 waves; block tile 64x64; wave computes 16 rows x 64 cols.
// LDS tiles are stored pre-swizzled into WMMA fragment order so each
// fragment loads as two ds_load_b128.
// ---------------------------------------------------------------------------
__global__ __launch_bounds__(128)
void k_gemm_wmma(const float* __restrict__ A, const float* __restrict__ Bw,
                 float* __restrict__ C,
                 int M, int N, int K, int lda, int ldb, int ldc,
                 long long sA, long long sB, long long sC,
                 const float* __restrict__ bias, int activ)
{
    // A: row-major [m][perm(k)], perm swaps k bits 3<->4 so that a lane's
    // 16 fragment halves are contiguous. B: transposed [n][perm(k)], same perm.
    __shared__ _Float16 lds_a[64][40];
    __shared__ _Float16 lds_b[64][40];

    const int bz = blockIdx.z;
    const float* Ab  = A + (size_t)bz * sA;
    const float* Bb_ = Bw + (size_t)bz * sB;
    float* Cb = C + (size_t)bz * sC;

    const int m0 = blockIdx.x * 64;
    const int n0 = blockIdx.y * 64;
    const int tid   = threadIdx.x;
    const int lane  = tid & 31;
    const int wv    = tid >> 5;
    const int mrow  = lane & 15;
    const int khalf = lane >> 4;

    v8f acc[4];
    #pragma unroll
    for (int t = 0; t < 4; ++t) acc[t] = (v8f){0.f,0.f,0.f,0.f,0.f,0.f,0.f,0.f};

    const int ksteps = (K + 31) / 32;
    for (int ks = 0; ks < ksteps; ++ks) {
        const int k0 = ks * 32;

        if (ks + 1 < ksteps) {   // speculative prefetch of next K tile
            __builtin_prefetch(Ab + (size_t)(m0 + (tid >> 2)) * lda + k0 + 32, 0, 1);
            __builtin_prefetch(Bb_ + (size_t)(k0 + 32 + (tid >> 4)) * ldb + n0 + ((tid & 15) << 2), 0, 1);
        }

        // ---- stage A tile 64x32 (f32 -> f16, fragment-permuted) ----
        #pragma unroll
        for (int i = 0; i < 4; ++i) {
            int idx = tid + i * 128;          // 0..511 float4 slots
            int r   = idx >> 3;               // tile row 0..63
            int c4  = (idx & 7) << 2;         // k offset 0,4,...,28
            int gm  = m0 + r;
            float f0, f1, f2, f3;
            if (gm < M && (k0 + 31) < K) {
                const float4 q = *(const float4*)(Ab + (size_t)gm * lda + k0 + c4);
                f0 = q.x; f1 = q.y; f2 = q.z; f3 = q.w;
            } else {
                const float* p = Ab + (size_t)gm * lda + k0 + c4;
                bool rok = gm < M;
                f0 = (rok && k0 + c4 + 0 < K) ? p[0] : 0.f;
                f1 = (rok && k0 + c4 + 1 < K) ? p[1] : 0.f;
                f2 = (rok && k0 + c4 + 2 < K) ? p[2] : 0.f;
                f3 = (rok && k0 + c4 + 3 < K) ? p[3] : 0.f;
            }
            int pb = (c4 & 7) | ((c4 & 8) << 1) | ((c4 & 16) >> 1);  // swap bits 3,4
            v4h hv = {(_Float16)f0, (_Float16)f1, (_Float16)f2, (_Float16)f3};
            *(v4h*)&lds_a[r][pb] = hv;
        }
        // ---- stage B tile 32x64 transposed to [n][perm(k)] ----
        #pragma unroll
        for (int i = 0; i < 4; ++i) {
            int idx = tid + i * 128;          // 0..511
            int n   = idx & 63;               // local col
            int kg  = (idx >> 6) << 2;        // k group 0,4,...,28
            int gn  = n0 + n;
            float f0, f1, f2, f3;
            if ((k0 + 31) < K && gn < N) {
                const float* p = Bb_ + (size_t)(k0 + kg) * ldb + gn;
                f0 = p[0];
                f1 = p[(size_t)ldb];
                f2 = p[(size_t)ldb * 2];
                f3 = p[(size_t)ldb * 3];
            } else {
                const float* p = Bb_ + (size_t)(k0 + kg) * ldb + gn;
                bool cok = gn < N;
                f0 = (cok && k0 + kg + 0 < K) ? p[0] : 0.f;
                f1 = (cok && k0 + kg + 1 < K) ? p[(size_t)ldb] : 0.f;
                f2 = (cok && k0 + kg + 2 < K) ? p[(size_t)ldb * 2] : 0.f;
                f3 = (cok && k0 + kg + 3 < K) ? p[(size_t)ldb * 3] : 0.f;
            }
            v4h hv = {(_Float16)f0, (_Float16)f1, (_Float16)f2, (_Float16)f3};
            *(v4h*)&lds_b[n][(kg & 7) | ((kg & 8) << 1) | ((kg & 16) >> 1)] = hv;
        }
        __syncthreads();

        // ---- fragments + WMMA ----
        const _Float16* ap = &lds_a[wv * 16 + mrow][khalf * 16];
        v8h alo = *(const v8h*)ap;
        v8h ahi = *(const v8h*)(ap + 8);
        v16h af = __builtin_shufflevector(alo, ahi,
                    0,1,2,3,4,5,6,7,8,9,10,11,12,13,14,15);
        #pragma unroll
        for (int t = 0; t < 4; ++t) {
            const _Float16* bp = &lds_b[t * 16 + mrow][khalf * 16];
            v8h blo = *(const v8h*)bp;
            v8h bhi = *(const v8h*)(bp + 8);
            v16h bf = __builtin_shufflevector(blo, bhi,
                        0,1,2,3,4,5,6,7,8,9,10,11,12,13,14,15);
            acc[t] = __builtin_amdgcn_wmma_f32_16x16x32_f16(
                false, af, false, bf, (short)0, acc[t], false, false);
        }
        __syncthreads();
    }

    // ---- epilogue: optional bias + tanh, store ----
    #pragma unroll
    for (int t = 0; t < 4; ++t) {
        #pragma unroll
        for (int r = 0; r < 8; ++r) {
            int gm = m0 + wv * 16 + khalf * 8 + r;
            int gn = n0 + t * 16 + mrow;
            if (gm < M && gn < N) {
                float vv = acc[t][r];
                if (bias) vv += bias[gn];
                if (activ) vv = tanhf(vv);
                Cb[(size_t)gm * ldc + gn] = vv;
            }
        }
    }
}

// ---------------------------------------------------------------------------
// Graph pooling: h_pooled[b,h] = sum_n graph_pool[b,n] * h[b,n,h]
// ---------------------------------------------------------------------------
__global__ void k_hpool(const float* __restrict__ gp, const float* __restrict__ h,
                        float* __restrict__ hp)
{
    const int b = blockIdx.x, t = threadIdx.x;
    float acc = 0.f;
    const float* gpb = gp + (size_t)b * NN;
    const float* hb  = h + (size_t)b * NN * HH + t;
    for (int n = 0; n < NN; ++n) acc += gpb[n] * hb[(size_t)n * HH];
    hp[(size_t)b * HH + t] = acc;
}

// ---------------------------------------------------------------------------
// Build actor concat matrix [B*NJ, 3H] = [cand_feat | h_pooled | mch_pool]
// ---------------------------------------------------------------------------
__global__ void k_build_concat(const int* __restrict__ candidate,
                               const float* __restrict__ hnodes,
                               const float* __restrict__ hp,
                               const float* __restrict__ mch_pool,
                               float* __restrict__ concat)
{
    const int row = blockIdx.x;              // 0..1919
    const int b = row / NJ, j = row % NJ;
    const int t = threadIdx.x;
    const int cand = candidate[b * NJ + j];
    float* cr = concat + (size_t)row * (3 * HH);
    cr[t]            = hnodes[((size_t)b * NN + cand) * HH + t];
    cr[HH + t]       = hp[(size_t)b * HH + t];
    cr[2 * HH + t]   = mch_pool[(size_t)b * HH + t];
}

// output layout (floats)
#define OUT_ENT   0
#define OUT_V     64
#define OUT_LOGA  128
#define OUT_ANODE 192
#define OUT_AFEAT 1472
#define OUT_MMCH  9664
#define OUT_HPOOL 10944

// ---------------------------------------------------------------------------
// Final actor layer (H->1, *10), masked log-softmax, entropy, log_a + critic.
// One block (128 threads) per batch element.
// ---------------------------------------------------------------------------
__global__ __launch_bounds__(128)
void k_actor_final(const float* __restrict__ act2,          // [B*NJ, H] (tanh'd)
                   const unsigned char* __restrict__ mask,  // [B,NJ] bool
                   const int* __restrict__ a_index,         // [B]
                   const float* __restrict__ hp,            // [B,H]
                   const float* __restrict__ Wa3, const float* __restrict__ ba3,
                   const float* __restrict__ Wc1, const float* __restrict__ bc1,
                   const float* __restrict__ Wc2, const float* __restrict__ bc2,
                   float* __restrict__ out)
{
    __shared__ float shp[HH], red[HH], sc[NJ];
    const int b = blockIdx.x, t = threadIdx.x;
    shp[t] = hp[(size_t)b * HH + t];
    const float w3 = Wa3[t];
    __syncthreads();

    for (int j = 0; j < NJ; ++j) {
        red[t] = act2[((size_t)b * NJ + j) * HH + t] * w3;
        __syncthreads();
        for (int s = 64; s > 0; s >>= 1) {
            if (t < s) red[t] += red[t + s];
            __syncthreads();
        }
        if (t == 0) sc[j] = (red[0] + ba3[0]) * 10.0f;
        __syncthreads();
    }

    if (t == 0) {
        const unsigned char* mk = mask + b * NJ;
        float m = -__builtin_huge_valf();
        for (int j = 0; j < NJ; ++j) if (!mk[j] && sc[j] > m) m = sc[j];
        float se = 0.f;
        for (int j = 0; j < NJ; ++j) if (!mk[j]) se += expf(sc[j] - m);
        float logZ = m + logf(se);
        float ent = 0.f;
        for (int j = 0; j < NJ; ++j)
            if (!mk[j]) { float lp = sc[j] - logZ; ent -= expf(lp) * lp; }
        out[OUT_ENT + b]  = ent;
        out[OUT_LOGA + b] = sc[a_index[b]] - logZ;
    }
    __syncthreads();

    // critic: v = tanh(hp @ Wc1 + bc1) @ Wc2 + bc2
    float accv = bc1[t];
    for (int k = 0; k < HH; ++k) accv += shp[k] * Wc1[(size_t)k * HH + t];
    red[t] = tanhf(accv) * Wc2[t];
    __syncthreads();
    for (int s = 64; s > 0; s >>= 1) {
        if (t < s) red[t] += red[t + s];
        __syncthreads();
    }
    if (t == 0) out[OUT_V + b] = red[0] + bc2[0];
}

// ---------------------------------------------------------------------------
// Action-dependent gathers + h_pooled copy-out
// ---------------------------------------------------------------------------
__global__ void k_gather(const int* __restrict__ old_action,
                         const float* __restrict__ dur,              // [B,N,NM]
                         const unsigned char* __restrict__ mask_mch, // [B,N,NM]
                         const float* __restrict__ hnodes,
                         const float* __restrict__ hp,
                         float* __restrict__ out)
{
    const int b = blockIdx.x, t = threadIdx.x;
    const int oa = old_action[b];
    if (t < NM) {
        out[OUT_ANODE + b * NM + t] = dur[((size_t)b * NN + oa) * NM + t];
        out[OUT_MMCH + b * NM + t]  = (float)mask_mch[((size_t)b * NN + oa) * NM + t];
    }
    out[OUT_AFEAT + b * HH + t] = hnodes[((size_t)b * NN + oa) * HH + t];
    out[OUT_HPOOL + b * HH + t] = hp[(size_t)b * HH + t];
}

// ---------------------------------------------------------------------------
extern "C" void kernel_launch(void* const* d_in, const int* in_sizes, int n_in,
                              void* d_out, int out_size, void* d_ws, size_t ws_size,
                              hipStream_t stream) {
    const float* x          = (const float*)d_in[0];
    const float* graph_pool = (const float*)d_in[1];
    const float* adj        = (const float*)d_in[3];
    const int*   candidate  = (const int*)d_in[4];
    const unsigned char* mask     = (const unsigned char*)d_in[5];
    const unsigned char* mask_mch = (const unsigned char*)d_in[6];
    const float* dur        = (const float*)d_in[7];
    const int*   a_index    = (const int*)d_in[8];
    const int*   old_action = (const int*)d_in[9];
    const float* mch_pool   = (const float*)d_in[10];
    const float* W1_0 = (const float*)d_in[11];
    const float* W2_0 = (const float*)d_in[13];
    const float* W1_1 = (const float*)d_in[15];
    const float* W2_1 = (const float*)d_in[17];
    const float* Wa1 = (const float*)d_in[19]; const float* ba1 = (const float*)d_in[20];
    const float* Wa2 = (const float*)d_in[21]; const float* ba2 = (const float*)d_in[22];
    const float* Wa3 = (const float*)d_in[23]; const float* ba3 = (const float*)d_in[24];
    const float* Wc1 = (const float*)d_in[25]; const float* bc1 = (const float*)d_in[26];
    const float* Wc2 = (const float*)d_in[27]; const float* bc2 = (const float*)d_in[28];
    float* out = (float*)d_out;

    float* bufA   = (float*)d_ws;                                  // [38400,128]
    float* bufB   = bufA + (size_t)RROWS * HH;                     // [38400,128]
    float* stats  = bufB + (size_t)RROWS * HH;                     // [256]
    float* hpool  = stats + 256;                                   // [64,128]
    float* concat = hpool + (size_t)Bb * HH;                       // [1920,384]
    float* act1   = concat + (size_t)AROWS * 3 * HH;               // [1920,128]
    float* act2   = act1 + (size_t)AROWS * HH;                     // [1920,128]

    dim3 blk128(128), blk256(256);
    dim3 g_mlp(RROWS / 64, HH / 64, 1);           // (600, 2, 1)
    dim3 g_adj((NN + 63) / 64, HH / 64, Bb);      // (10, 2, 64)
    dim3 g_a1(AROWS / 64, HH / 64, 1);            // (30, 2, 1)

    // --- GIN layer 1 ---
    k_layer1<<<RROWS, blk128, 0, stream>>>(adj, x, W1_0, bufB);
    k_zero_stats<<<1, blk256, 0, stream>>>(stats);
    k_stats<<<256, blk256, 0, stream>>>(bufB, stats);
    k_norm_relu<<<4800, blk256, 0, stream>>>(bufB, stats);        // h1 in bufB

    k_gemm_wmma<<<g_mlp, blk128, 0, stream>>>(bufB, W2_0, bufA,
        RROWS, HH, HH, HH, HH, HH, 0, 0, 0, nullptr, 0);
    k_zero_stats<<<1, blk256, 0, stream>>>(stats);
    k_stats<<<256, blk256, 0, stream>>>(bufA, stats);
    k_norm_relu<<<4800, blk256, 0, stream>>>(bufA, stats);        // h (L1 out) in bufA

    // --- GIN layer 2 ---
    k_gemm_wmma<<<g_adj, blk128, 0, stream>>>(adj, bufA, bufB,
        NN, HH, NN, NN, HH, HH,
        (long long)NN * NN, (long long)NN * HH, (long long)NN * HH,
        nullptr, 0);                                              // pooled in bufB

    k_gemm_wmma<<<g_mlp, blk128, 0, stream>>>(bufB, W1_1, bufA,
        RROWS, HH, HH, HH, HH, HH, 0, 0, 0, nullptr, 0);
    k_zero_stats<<<1, blk256, 0, stream>>>(stats);
    k_stats<<<256, blk256, 0, stream>>>(bufA, stats);
    k_norm_relu<<<4800, blk256, 0, stream>>>(bufA, stats);        // h1' in bufA

    k_gemm_wmma<<<g_mlp, blk128, 0, stream>>>(bufA, W2_1, bufB,
        RROWS, HH, HH, HH, HH, HH, 0, 0, 0, nullptr, 0);
    k_zero_stats<<<1, blk256, 0, stream>>>(stats);
    k_stats<<<256, blk256, 0, stream>>>(bufB, stats);
    k_norm_relu<<<4800, blk256, 0, stream>>>(bufB, stats);        // h_nodes in bufB

    // --- pooling ---
    k_hpool<<<Bb, blk128, 0, stream>>>(graph_pool, bufB, hpool);

    // --- actor MLP as WMMA GEMMs with fused bias+tanh ---
    k_build_concat<<<AROWS, blk128, 0, stream>>>(candidate, bufB, hpool,
                                                 mch_pool, concat);
    k_gemm_wmma<<<g_a1, blk128, 0, stream>>>(concat, Wa1, act1,
        AROWS, HH, 3 * HH, 3 * HH, HH, HH, 0, 0, 0, ba1, 1);
    k_gemm_wmma<<<g_a1, blk128, 0, stream>>>(act1, Wa2, act2,
        AROWS, HH, HH, HH, HH, HH, 0, 0, 0, ba2, 1);

    // --- heads ---
    k_actor_final<<<Bb, blk128, 0, stream>>>(act2, mask, a_index, hpool,
        Wa3, ba3, Wc1, bc1, Wc2, bc2, out);
    k_gather<<<Bb, blk128, 0, stream>>>(old_action, dur, mask_mch, bufB, hpool, out);
}
